// _PointnetSAModuleBase_att_69561290326464
// MI455X (gfx1250) — matrix operand order repaired and checked by reading, not compiled, not run on hardware
//
#include <hip/hip_runtime.h>
#include <hip/hip_bf16.h>
#include <math.h>

// ---------------------------------------------------------------------------
// Types for CDNA5 WMMA (wave32): D(16x16 f32) = A(16x32 bf16) x B(32x16 bf16) + C
// ---------------------------------------------------------------------------
typedef __attribute__((ext_vector_type(16))) __bf16 v16bf;
typedef __attribute__((ext_vector_type(8)))  __bf16 v8bf;
typedef __attribute__((ext_vector_type(8)))  float  v8f;

#define WMMA_BF16(a, b, c) \
    __builtin_amdgcn_wmma_f32_16x16x32_bf16(false, (a), false, (b), (short)0, (c), false, false)

// A-matrix fragment (16x32 bf16) from row-major f32 weights W[COUT][CIN].
// Lane l holds row M = base + (l & 15); element e -> K = kb + half*16 + e.
template<int CIN>
static __device__ inline v16bf load_afrag(const float* __restrict__ W, int row, int kb, int hl) {
    const float* wr = W + row * CIN;
    v16bf a;
#pragma unroll
    for (int e = 0; e < 16; ++e) {
        int K = kb + (hl << 4) + e;
        float v = (K < CIN) ? wr[K] : 0.0f;
        a[e] = (__bf16)v;
    }
    return a;
}

// B-matrix fragment (32x16 bf16) from LDS stored point-major: X[point][channel], stride kp.
// Lane<16: col=lane, K=kb+e (contiguous); lane>=16: col=lane-16, K=kb+16+e (contiguous).
// -> one aligned 32-byte read per lane (two 16B LDS loads).
static __device__ inline v16bf load_bfrag(const __bf16* X, int kp, int ntile, int lane, int kb) {
    int col = lane & 15, hl = lane >> 4;
    const __bf16* p = X + (ntile * 16 + col) * kp + kb + hl * 16;
    v8bf lo = *(const v8bf*)(p);
    v8bf hi = *(const v8bf*)(p + 8);
    return __builtin_shufflevector(lo, hi, 0, 1, 2, 3, 4, 5, 6, 7, 8, 9, 10, 11, 12, 13, 14, 15);
}

// ---------------------------------------------------------------------------
// Kernel 1: Furthest point sampling. One block per batch. Writes new_xyz (B,S,3).
// ---------------------------------------------------------------------------
#define FPS_T 256
__global__ __launch_bounds__(FPS_T) void fps_kernel(const float* __restrict__ xyz,
                                                    float* __restrict__ out_newxyz) {
    __shared__ float sx[4096], sy[4096], sz[4096];
    __shared__ float rv[8];
    __shared__ int   ri[8];
    __shared__ int   curS;

    const int b = blockIdx.x, t = threadIdx.x;
    const int lane = t & 31, wid = t >> 5;
    const float* base = xyz + b * 4096 * 3;
    float* onx = out_newxyz + b * 1024 * 3;

    for (int p = t; p < 4096; p += FPS_T) {
        sx[p] = base[p * 3 + 0];
        sy[p] = base[p * 3 + 1];
        sz[p] = base[p * 3 + 2];
    }
    float mind[16];
#pragma unroll
    for (int j = 0; j < 16; ++j) mind[j] = 1e10f;
    if (t == 0) {
        onx[0] = base[0]; onx[1] = base[1]; onx[2] = base[2];
        curS = 0;
    }
    __syncthreads();

    int cur = 0;
    for (int i = 1; i < 1024; ++i) {
        float lx = sx[cur], ly = sy[cur], lz = sz[cur];
        float bv = -1.0f; int bi = 0x7fffffff;
#pragma unroll
        for (int j = 0; j < 16; ++j) {
            int p = t + j * FPS_T;
            float dx = sx[p] - lx, dy = sy[p] - ly, dz = sz[p] - lz;
            float d = dx * dx + dy * dy + dz * dz;
            d = fminf(d, mind[j]);
            mind[j] = d;
            if (d > bv || (d == bv && p < bi)) { bv = d; bi = p; }
        }
        // wave32 argmax reduce (first index wins ties, to match jnp.argmax)
        for (int off = 16; off; off >>= 1) {
            float ov = __shfl_xor(bv, off);
            int   oi = __shfl_xor(bi, off);
            if (ov > bv || (ov == bv && oi < bi)) { bv = ov; bi = oi; }
        }
        if (lane == 0) { rv[wid] = bv; ri[wid] = bi; }
        __syncthreads();
        if (wid == 0) {
            float v = (lane < 8) ? rv[lane] : -1.0f;
            int  ix = (lane < 8) ? ri[lane] : 0x7fffffff;
            for (int off = 4; off; off >>= 1) {
                float ov = __shfl_xor(v, off);
                int   oi = __shfl_xor(ix, off);
                if (ov > v || (ov == v && oi < ix)) { v = ov; ix = oi; }
            }
            if (lane == 0) {
                curS = ix;
                onx[i * 3 + 0] = sx[ix];
                onx[i * 3 + 1] = sy[ix];
                onx[i * 3 + 2] = sz[ix];
            }
        }
        __syncthreads();
        cur = curS;
    }
}

// ---------------------------------------------------------------------------
// Kernel 2: attention gate. att = sigmoid(att_in @ w_att + b_att), att_in (16,3072).
// 16 blocks x 1 wave; block j computes the 16x16 output tile for channels [16j,16j+16).
// ---------------------------------------------------------------------------
__global__ __launch_bounds__(32) void att_kernel(const float* __restrict__ newxyz,
                                                 const float* __restrict__ w_att,
                                                 const float* __restrict__ b_att,
                                                 float* __restrict__ att) {
    const int j = blockIdx.x;
    const int lane = threadIdx.x & 31;
    const int hl = lane >> 4, lr = lane & 15;
    const float* arow = newxyz + lr * 3072;  // batch row lr, (3,S) flattened view of (S,3)

    v8f acc = {0.f, 0.f, 0.f, 0.f, 0.f, 0.f, 0.f, 0.f};
    for (int kb = 0; kb < 3072; kb += 32) {
        v16bf a, w;
#pragma unroll
        for (int e = 0; e < 16; ++e) {
            int Ka = kb + (hl << 4) + e;
            int dim = Ka >> 10, s = Ka & 1023;          // att_in[b, dim*1024+s] = new_xyz[b,s,dim]
            a[e] = (__bf16)arow[s * 3 + dim];
            int Kb = kb + hl * 16 + e;
            w[e] = (__bf16)w_att[Kb * 256 + j * 16 + lr];
        }
        acc = WMMA_BF16(a, w, acc);
    }
#pragma unroll
    for (int r = 0; r < 8; ++r) {
        int M = r + 8 * hl;       // batch
        int ch = j * 16 + lr;     // channel
        float x = acc[r] + b_att[ch];
        att[M * 256 + ch] = 1.0f / (1.0f + __expf(-x));
    }
}

// ---------------------------------------------------------------------------
// MLP layer (64 outputs): Y[point][ch] = relu(W @ X + b), bf16 LDS -> bf16 LDS.
// ---------------------------------------------------------------------------
template<int CIN, int KPIN, int KPOUT>
static __device__ inline void mlp_layer64(const __bf16* Xl, __bf16* Yl,
                                          const float* __restrict__ W,
                                          const float* __restrict__ bias,
                                          int lane, int wave) {
    constexpr int KT = (CIN + 31) / 32;
    const int col = lane & 15, hl = lane >> 4;
#pragma unroll
    for (int mt = 0; mt < 4; ++mt) {
        v8f a0 = {0.f, 0.f, 0.f, 0.f, 0.f, 0.f, 0.f, 0.f};
        v8f a1 = a0;
#pragma unroll
        for (int kt = 0; kt < KT; ++kt) {
            int kb = kt * 32;
            v16bf af = load_afrag<CIN>(W, mt * 16 + col, kb, hl);
            v16bf b0 = load_bfrag(Xl, KPIN, wave * 2 + 0, lane, kb);
            v16bf b1 = load_bfrag(Xl, KPIN, wave * 2 + 1, lane, kb);
            a0 = WMMA_BF16(af, b0, a0);
            a1 = WMMA_BF16(af, b1, a1);
        }
        int n0 = wave * 32 + col;
#pragma unroll
        for (int r = 0; r < 8; ++r) {
            int M = mt * 16 + r + 8 * hl;
            float bia = bias[M];
            Yl[n0 * KPOUT + M]        = (__bf16)fmaxf(a0[r] + bia, 0.0f);
            Yl[(n0 + 16) * KPOUT + M] = (__bf16)fmaxf(a1[r] + bia, 0.0f);
        }
    }
}

// ---------------------------------------------------------------------------
// Kernel 3: ball query + group + MLP(WMMA) + maxpool + att scale, fused.
// 128 threads = 4 waves; each workgroup handles TC = 128/NS centers (128 grouped pts).
// ---------------------------------------------------------------------------
template<int NS>
__global__ __launch_bounds__(128) void group_mlp_kernel(
    const float* __restrict__ xyz, const float* __restrict__ feats,
    const float* __restrict__ newxyz, const float* __restrict__ att,
    const float* __restrict__ w1, const float* __restrict__ b1,
    const float* __restrict__ w2, const float* __restrict__ b2,
    const float* __restrict__ w3, const float* __restrict__ b3,
    float* __restrict__ outF, float r2, int cbase) {
    constexpr int TC = 128 / NS;      // centers per workgroup
    constexpr int KP1 = 104;          // layer-1 input stride (67 padded to 96, +bank pad)
    constexpr int KP2 = 72;           // 64-channel stride (+bank pad)

    __shared__ __bf16 Xb[128 * KP1];
    __shared__ __bf16 Yb[128 * KP2];
    __shared__ float  sC[TC * 3];
    __shared__ int    ballIdx[128];
    __shared__ float  pmax[4][128];

    const int tilesPerB = 1024 / TC;
    const int b  = blockIdx.x / tilesPerB;
    const int s0 = (blockIdx.x % tilesPerB) * TC;
    const int t = threadIdx.x, lane = t & 31, wave = t >> 5;
    const int hl = lane >> 4, col = lane & 15;

    if (t < TC * 3) sC[t] = newxyz[b * 3072 + (s0 + t / 3) * 3 + (t % 3)];
    __syncthreads();

    // ---- ball query: one wave per center, ordered scan (first NS hits by index) ----
    if (wave < TC) {
        const int c = wave;
        const float cx = sC[c * 3 + 0], cy = sC[c * 3 + 1], cz = sC[c * 3 + 2];
        const float* xb = xyz + b * 4096 * 3;
        int cnt = 0, first = -1;
        for (int bs = 0; bs < 4096 && cnt < NS; bs += 32) {
            int p = bs + lane;
            float dx = xb[p * 3 + 0] - cx;
            float dy = xb[p * 3 + 1] - cy;
            float dz = xb[p * 3 + 2] - cz;
            float d2 = dx * dx + dy * dy + dz * dz;
            unsigned m = (unsigned)__ballot(d2 < r2);
            if (first < 0 && m) first = bs + __ffs(m) - 1;
            bool in = (m >> lane) & 1u;
            int pre = __popc(m & ((1u << lane) - 1u));
            int slot = cnt + pre;
            if (in && slot < NS) ballIdx[c * NS + slot] = p;
            cnt += __popc(m);
        }
        if (cnt > NS) cnt = NS;
        int fidx = (first < 0) ? 0 : first;  // pad empty slots with first hit (ref semantics)
        for (int jj = cnt + lane; jj < NS; jj += 32) ballIdx[c * NS + jj] = fidx;
    }
    __syncthreads();

    // ---- gather: X[point][0..2]=rel xyz, [3..66]=features, [67..95]=0 (bf16) ----
    for (int n = wave; n < 128; n += 4) {
        int idx = ballIdx[n];
        int c = n / NS;
        const float* q = xyz + (b * 4096 + idx) * 3;
        const float* f = feats + (b * 4096 + idx) * 64;
        float v0 = (lane < 3) ? (q[lane] - sC[c * 3 + lane]) : f[lane - 3];
        Xb[n * KP1 + lane] = (__bf16)v0;
        Xb[n * KP1 + 32 + lane] = (__bf16)f[lane + 29];
        float v2 = (lane < 3) ? f[61 + lane] : 0.0f;
        Xb[n * KP1 + 64 + lane] = (__bf16)v2;
    }
    __syncthreads();

    // ---- layer 1: 67 -> 64 (X -> Y), layer 2: 64 -> 64 (Y -> X) ----
    mlp_layer64<67, KP1, KP2>(Xb, Yb, w1, b1, lane, wave);
    __syncthreads();
    mlp_layer64<64, KP2, KP1>(Yb, Xb, w2, b2, lane, wave);
    __syncthreads();

    // ---- layer 3: 64 -> 128, fused in-register maxpool over this wave's 32 points ----
#pragma unroll
    for (int mt = 0; mt < 8; ++mt) {
        v8f a0 = {0.f, 0.f, 0.f, 0.f, 0.f, 0.f, 0.f, 0.f};
        v8f a1 = a0;
#pragma unroll
        for (int kt = 0; kt < 2; ++kt) {
            int kb = kt * 32;
            v16bf af = load_afrag<64>(w3, mt * 16 + col, kb, hl);
            v16bf b0 = load_bfrag(Xb, KP1, wave * 2 + 0, lane, kb);
            v16bf b1 = load_bfrag(Xb, KP1, wave * 2 + 1, lane, kb);
            a0 = WMMA_BF16(af, b0, a0);
            a1 = WMMA_BF16(af, b1, a1);
        }
        float m8[8];
#pragma unroll
        for (int r = 0; r < 8; ++r) m8[r] = fmaxf(a0[r], a1[r]);
#pragma unroll
        for (int off = 1; off <= 8; off <<= 1) {
#pragma unroll
            for (int r = 0; r < 8; ++r) m8[r] = fmaxf(m8[r], __shfl_xor(m8[r], off));
        }
        if (col == 0) {
#pragma unroll
            for (int r = 0; r < 8; ++r) {
                int M = mt * 16 + r + 8 * hl;
                pmax[wave][M] = fmaxf(m8[r] + b3[M], 0.0f);  // relu(max+bias) == max(relu(x+bias))
            }
        }
    }
    __syncthreads();

    // ---- combine partials across waves of a center, scale by attention, store ----
    {
        constexpr int WPC = NS / 32;  // waves per center
        int ch = t;                   // 0..127
#pragma unroll
        for (int c = 0; c < TC; ++c) {
            float m = pmax[c * WPC][ch];
#pragma unroll
            for (int w = 1; w < WPC; ++w) m = fmaxf(m, pmax[c * WPC + w][ch]);
            outF[(b * 256 + cbase + ch) * 1024 + s0 + c] = m * att[b * 256 + cbase + ch];
        }
    }
}

// ---------------------------------------------------------------------------
// Launch
// ---------------------------------------------------------------------------
extern "C" void kernel_launch(void* const* d_in, const int* in_sizes, int n_in,
                              void* d_out, int out_size, void* d_ws, size_t ws_size,
                              hipStream_t stream) {
    const float* xyz   = (const float*)d_in[0];
    const float* feats = (const float*)d_in[1];
    const float* w_att = (const float*)d_in[2];
    const float* b_att = (const float*)d_in[3];
    const float* w00 = (const float*)d_in[4],  *b00 = (const float*)d_in[5];
    const float* w01 = (const float*)d_in[6],  *b01 = (const float*)d_in[7];
    const float* w02 = (const float*)d_in[8],  *b02 = (const float*)d_in[9];
    const float* w10 = (const float*)d_in[10], *b10 = (const float*)d_in[11];
    const float* w11 = (const float*)d_in[12], *b11 = (const float*)d_in[13];
    const float* w12 = (const float*)d_in[14], *b12 = (const float*)d_in[15];

    float* out_newxyz = (float*)d_out;                 // (16, 1024, 3)
    float* out_feat   = out_newxyz + 16 * 1024 * 3;    // (16, 256, 1024)
    float* att        = (float*)d_ws;                  // (16, 256) scratch

    fps_kernel<<<16, FPS_T, 0, stream>>>(xyz, out_newxyz);
    att_kernel<<<16, 32, 0, stream>>>(out_newxyz, w_att, b_att, att);
    // group 0: radius 0.2, ns=32, channels [0,128)
    group_mlp_kernel<32><<<16 * 256, 128, 0, stream>>>(xyz, feats, out_newxyz, att,
                                                       w00, b00, w01, b01, w02, b02,
                                                       out_feat, 0.04f, 0);
    // group 1: radius 0.4, ns=64, channels [128,256)
    group_mlp_kernel<64><<<16 * 512, 128, 0, stream>>>(xyz, feats, out_newxyz, att,
                                                       w10, b10, w11, b11, w12, b12,
                                                       out_feat, 0.16f, 128);
}
